// ConvGRU_64845416235161
// MI455X (gfx1250) — matrix operand order repaired and checked by reading, not compiled, not run on hardware
//
#include <hip/hip_runtime.h>
#include <hip/hip_bf16.h>
#include <math.h>

typedef __attribute__((ext_vector_type(16))) _Float16 v16h;
typedef __attribute__((ext_vector_type(8)))  _Float16 v8h;
typedef __attribute__((ext_vector_type(8)))  float    v8f;
typedef __attribute__((ext_vector_type(4)))  unsigned int v4u;
typedef __attribute__((ext_vector_type(8)))  int v8i;
typedef __attribute__((ext_vector_type(4)))  int v4i;

#define Bv   4
#define Tv   16
#define CIN  32
#define CH   64
#define HW   4096       // 64*64
#define NW   192        // 3*hidden
#define NCOL 256        // fused gate columns: [r | z | i_n | h_n]
#define MTOT (Bv*HW)    // 16384

// channel stride in LDS (f16 elems): 40 -> 80B stride -> conflict-free b128 reads
#define CSP  40
#define XW   66         // haloed x extent: x in [-1, 64] -> xr = x+1 in [0,65]

// one weight LDS image: [9 taps][64 n][CSP c] f16
#define WIMG_ELEMS (9 * 64 * CSP)          // 23040 f16
#define WIMG_BYTES (WIMG_ELEMS * 2)        // 46080 B = 5760 x 8B
#define WIMG_D8    (WIMG_BYTES / 8)        // 5760 data_size=8B elements

__device__ __forceinline__ float sigmoidf_(float x) { return 1.0f / (1.0f + __expf(-x)); }

// TDM issue: D# groups per ISA 8.3/8.4; arity differs between clang-22 (ROCm 7.2,
// 5 args) and clang-23 (amdgpu-toolchain, 6 args) per probe results.
__device__ __forceinline__ void tdm_copy_to_lds(unsigned lds_off, unsigned long long ga) {
  v4u g0;
  g0[0] = 1u;                                        // count=1, no gather
  g0[1] = lds_off;                                   // lds_addr
  g0[2] = (unsigned)(ga & 0xFFFFFFFFull);            // global_addr[31:0]
  g0[3] = (unsigned)((ga >> 32) & 0x1FFFFFFull) | (2u << 30);  // ga[56:32] | type=2
  v8i g1;
  g1[0] = (int)(3u << 16);                           // wg_mask=0, data_size=8B
  g1[1] = (int)((WIMG_D8 & 0xFFFFu) << 16);          // tensor_dim0[15:0] @ bits 63:48
  g1[2] = (int)(1u << 16);                           // tensor_dim0[31:16]=0, tensor_dim1=1
  g1[3] = (int)((WIMG_D8 & 0xFFFFu) << 16);          // tile_dim0 @ bits 127:112
  g1[4] = 1;                                         // tile_dim1=1, tile_dim2=0
  g1[5] = (int)WIMG_D8;                              // tensor_dim0_stride[31:0]
  g1[6] = 0;
  g1[7] = 0;
  v4i z = {0, 0, 0, 0};
#if defined(__clang_major__) && (__clang_major__ >= 23)
  v8i z8 = {0, 0, 0, 0, 0, 0, 0, 0};
  __builtin_amdgcn_tensor_load_to_lds(g0, g1, z, z, z8, 0);
#else
  __builtin_amdgcn_tensor_load_to_lds(g0, g1, z, z, 0);
#endif
}

// ---- weight prep: build 9 padded LDS-image copies of the weights in f16 ----
// images 0..2:  Wi gate blocks (noff = img*64),           c-slice 0
// images 3..8:  Wh (j = (img-3)>>1 -> noff {0,64,128}),   c-slice cs = (img-3)&1
// image layout: [tap][n][CSP] with c in [0,32) valid, [32,40) pad (never read)
__global__ void prep_weights_kernel(const float* __restrict__ Wi,
                                    const float* __restrict__ Wh,
                                    _Float16* __restrict__ WImg) {
  int idx = blockIdx.x * blockDim.x + threadIdx.x;
  if (idx >= 9 * 9 * 64 * 32) return;
  int img = idx / 18432;            // 0..8
  int rem = idx % 18432;
  int tap = rem >> 11;              // 0..8
  int n   = (rem >> 5) & 63;
  int c   = rem & 31;
  float v;
  if (img < 3) {
    int noff = img * 64;
    v = Wi[((size_t)(noff + n) * CIN + c) * 9 + tap];
  } else {
    int j    = (img - 3) >> 1;
    int cs   = (img - 3) & 1;
    int noff = (j == 2) ? 128 : j * 64;
    v = Wh[((size_t)(noff + n) * CH + cs * 32 + c) * 9 + tap];
  }
  WImg[(size_t)img * WIMG_ELEMS + (tap * 64 + n) * CSP + c] = (_Float16)v;
}

// ---- fused gate GEMM for one timestep ----
// g[m][0:64)   = conv(x_t,Wi[0:64])   + conv(h,Wh[0:64])     (r pre-act)
// g[m][64:128) = conv(x_t,Wi[64:128]) + conv(h,Wh[64:128])   (z pre-act)
// g[m][128:192)= conv(x_t,Wi[128:192])                        (i_n)
// g[m][192:256)= conv(h,  Wh[128:192])                        (h_n)
__global__ __launch_bounds__(256)
void convgru_gemm_kernel(const float* __restrict__ x,      // [B,T,CIN,HW]
                         const float* __restrict__ out,    // [B,T,CH,HW] (h history)
                         const _Float16* __restrict__ WImg,// 9 x WIMG_ELEMS
                         float* __restrict__ g,            // [MTOT][NCOL]
                         int t) {
  // haloed input slice: [y 0..3][xr 0..65][c], c fastest (stride CSP)
  __shared__ __attribute__((aligned(16))) _Float16 Xs[4 * XW * CSP];   // 21120 B
  // weight image (TDM destination): [tap][n][CSP]
  __shared__ __attribute__((aligned(16))) _Float16 Bs[WIMG_ELEMS];     // 46080 B

  const int tid   = threadIdx.x;
  const int mb    = blockIdx.x;          // 0..127 : 128-row M tile
  const int nb    = blockIdx.y;          // 0..3   : 64-col gate block
  const int b     = mb >> 5;             // image
  const int ybase = (mb & 31) * 2;       // tile covers image rows ybase, ybase+1

  const int wave = tid >> 5;
  const int lane = tid & 31;
  const int wm   = wave >> 1;            // 0..3 : 32-row strip
  const int wn   = wave & 1;             // 0..1 : 32-col strip
  const int lrow = lane & 15;
  const int kh   = lane >> 4;            // K-half select per ISA 7.12.2 layout

  v8f acc[2][2] = {};

  const bool use_x = (nb != 3);
  const bool use_h = (nb != 2) && (t > 0);

  const float* xsrc = x + ((size_t)(b * Tv + t) * CIN) * HW;
  const float* hsrc = use_h ? (out + ((size_t)(b * Tv + (t - 1)) * CH) * HW) : nullptr;

  // one-time: halo columns xr=0 and xr=65 are OOB for every tap -> always zero
  {
    int c  = tid & 31;
    int xh = (tid >> 5) & 1;
    int y  = tid >> 6;                   // 0..3
    Xs[(y * XW + (xh ? 65 : 0)) * CSP + c] = (_Float16)0.0f;
  }

  // per-lane fragment row geometry (tap-independent)
  int ayl[2], axl[2];
  #pragma unroll
  for (int r = 0; r < 2; ++r) {
    int m  = wm * 32 + r * 16 + lrow;
    ayl[r] = m >> 6;                     // 0..1
    axl[r] = m & 63;                     // 0..63
  }

  const unsigned lds_bs = (unsigned)(unsigned long long)(const void*)&Bs[0];

  for (int src = 0; src < 2; ++src) {
    const bool en = (src == 0) ? use_x : use_h;
    if (!en) continue;
    const float* sp = (src == 0) ? xsrc : hsrc;
    const int nsteps = (src == 0) ? 1 : 2;

    for (int cs = 0; cs < nsteps; ++cs) {
      const int cbase = cs << 5;
      __syncthreads();   // previous tap-loop reads done before overwriting LDS

      // ---- TDM: pull this slice's padded weight image straight into LDS ----
      if (wave == 0) {   // wave-uniform branch (TDM ignores EXEC)
        int img = (src == 0) ? nb : (3 + (((nb == 3) ? 2 : nb) << 1) + cs);
        unsigned long long ga =
            (unsigned long long)(const void*)(WImg + (size_t)img * WIMG_ELEMS);
        tdm_copy_to_lds(lds_bs, ga);
        __builtin_amdgcn_s_wait_tensorcnt(0);
      }

      // ---- stage haloed X interior: 4y x 32c x 32 float2 = 16*256 pairs ----
      #pragma unroll 4
      for (int i = 0; i < 16; ++i) {
        int idx = tid + 256 * i;
        int xp  = idx & 31;              // xx = 2*xp
        int c   = (idx >> 5) & 31;
        int y   = idx >> 10;             // 0..3
        int yy  = ybase + y - 1;         // image row
        int inb = (unsigned)yy < 64u;
        size_t off = inb ? ((size_t)(cbase + c) * HW + (size_t)(yy * 64 + 2 * xp)) : 0;
        float2 v = *(const float2*)(sp + off);
        float v0 = inb ? v.x : 0.0f;
        float v1 = inb ? v.y : 0.0f;
        _Float16* dst = &Xs[(y * XW + 2 * xp + 1) * CSP + c];
        dst[0]   = (_Float16)v0;
        dst[CSP] = (_Float16)v1;
      }
      __syncthreads();   // TDM data + Xs stores visible to all waves

      // ---- 9 taps: pure LDS fragments + WMMA ----
      #pragma unroll
      for (int tap = 0; tap < 9; ++tap) {
        const int dy = tap / 3 - 1;
        const int dx = tap % 3 - 1;

        v16h aF[2], bF[2];
        #pragma unroll
        for (int r = 0; r < 2; ++r) {
          const int yl = ayl[r] + dy + 1;          // 0..3
          const int xr = axl[r] + dx + 1;          // 0..65
          const _Float16* ap = &Xs[(yl * XW + xr) * CSP];
          v8h lo = *(const v8h*)(ap + kh * 8);       // K kh*8 .. +7
          v8h hi = *(const v8h*)(ap + 16 + kh * 8);  // K 16+kh*8 .. +7
          #pragma unroll
          for (int e = 0; e < 8; ++e) { aF[r][e] = lo[e]; aF[r][8 + e] = hi[e]; }
        }
        #pragma unroll
        for (int q = 0; q < 2; ++q) {
          const int col = wn * 32 + q * 16 + lrow;
          const _Float16* bp = &Bs[(tap * 64 + col) * CSP];
          v8h lo = *(const v8h*)(bp + kh * 8);       // K kh*8 .. +7
          v8h hi = *(const v8h*)(bp + 16 + kh * 8);  // K 16+kh*8 .. +7
          #pragma unroll
          for (int e = 0; e < 8; ++e) { bF[q][e] = lo[e]; bF[q][8 + e] = hi[e]; }
        }
        #pragma unroll
        for (int r = 0; r < 2; ++r)
          #pragma unroll
          for (int q = 0; q < 2; ++q)
            acc[r][q] = __builtin_amdgcn_wmma_f32_16x16x32_f16(
                false, aF[r], false, bF[q], (short)0, acc[r][q], false, false);
      }
    }
  }

  // epilogue: C/D layout -> VGPR v, lane L holds (M = v + 8*(L>>4), N = L&15)
  const int mbase0 = mb * 128 + wm * 32;
  #pragma unroll
  for (int r = 0; r < 2; ++r) {
    #pragma unroll
    for (int q = 0; q < 2; ++q) {
      const int colg = nb * 64 + wn * 32 + q * 16 + lrow;
      const int mr   = mbase0 + r * 16 + kh * 8;
      #pragma unroll
      for (int v = 0; v < 8; ++v)
        g[(size_t)(mr + v) * NCOL + colg] = acc[r][q][v];
    }
  }
}

// ---- gate fusion + recurrence update (L2-resident) ----
__global__ __launch_bounds__(256)
void convgru_gate_kernel(const float* __restrict__ g,
                         float* __restrict__ out, int t) {
  int e = blockIdx.x * 256 + threadIdx.x;         // [B][CH][HW] = 1,048,576
  if (e >= Bv * CH * HW) return;
  int b   = e >> 18;                               // CH*HW = 2^18
  int rem = e & 262143;
  int nch = rem >> 12;
  int p   = rem & 4095;
  int m   = b * HW + p;
  const float* gr = g + (size_t)m * NCOL;
  float rg = sigmoidf_(gr[nch]);
  float z  = sigmoidf_(gr[64 + nch]);
  float nn = tanhf(gr[128 + nch] + rg * gr[192 + nch]);
  float hp = 0.0f;
  if (t > 0) hp = out[((size_t)(b * Tv + (t - 1)) * CH + nch) * HW + p];
  out[((size_t)(b * Tv + t) * CH + nch) * HW + p] = (1.0f - z) * nn + z * hp;
}

extern "C" void kernel_launch(void* const* d_in, const int* in_sizes, int n_in,
                              void* d_out, int out_size, void* d_ws, size_t ws_size,
                              hipStream_t stream) {
  const float* x  = (const float*)d_in[0];
  const float* Wi = (const float*)d_in[1];
  const float* Wh = (const float*)d_in[2];
  float* out = (float*)d_out;

  // ws layout: g (16 MiB f32) | 9 padded f16 weight images (415 KB)
  char* ws = (char*)d_ws;
  float*    g    = (float*)ws;
  _Float16* WImg = (_Float16*)(ws + (size_t)MTOT * NCOL * sizeof(float));

  {
    int total = 9 * 9 * 64 * 32;   // 165888
    prep_weights_kernel<<<(total + 255) / 256, 256, 0, stream>>>(Wi, Wh, WImg);
  }
  dim3 ggrid(128, 4);
  for (int t = 0; t < Tv; ++t) {
    convgru_gemm_kernel<<<ggrid, 256, 0, stream>>>(x, out, WImg, g, t);
    convgru_gate_kernel<<<4096, 256, 0, stream>>>(g, out, t);
  }
}
